// SelfAttentionLayer_89833535963909
// MI455X (gfx1250) — matrix operand order repaired and checked by reading, not compiled, not run on hardware
//
#include <hip/hip_runtime.h>

// ---------------------------------------------------------------------------
// SAGAN-style self-attention, fused flash-attention for MI455X (gfx1250).
//   x:  [8, 256, 64, 64] f32   (n = 4096 flattened spatial)
//   q,k = W[32,256]  @ x  -> bf16, stored transposed [b][n][32]
//   v   = W[256,256] @ x  -> bf16, stored [b][256][n]
//   out = gamma * softmax(q^T k) @ v + x
// Attention block: 32 queries x 256 channels, 8 waves; K/V tiles staged into
// LDS by the Tensor Data Mover (double buffered), matmuls via WMMA bf16.
// ---------------------------------------------------------------------------

typedef __attribute__((ext_vector_type(16))) __bf16 v16bf;
typedef __attribute__((ext_vector_type(8)))  __bf16 v8bf;
typedef __attribute__((ext_vector_type(8)))  float  v8f;
typedef __attribute__((ext_vector_type(8)))  unsigned int v8u;
typedef __attribute__((ext_vector_type(4)))  unsigned int v4u;
typedef __attribute__((ext_vector_type(8)))  int v8i;
typedef __attribute__((ext_vector_type(4)))  int v4i;

#define BATCH 8
#define CIN   256
#define NPIX  4096
#define DQK   32

// Two f32 -> packed bf16 pair in one dword: one v_add per input (round to
// nearest) + a single v_perm_b32 selecting the high halves of both words.
__device__ __forceinline__ unsigned pkbf(float a, float b) {
  unsigned au = __float_as_uint(a) + 0x8000u;
  unsigned bu = __float_as_uint(b) + 0x8000u;
  return __builtin_amdgcn_perm(bu, au, 0x07060302u);  // {hi16(b), hi16(a)}
}

// Scalar f32 -> bf16 (2 VALU ops)
__device__ __forceinline__ __bf16 f2bf(float f) {
  unsigned u = (__float_as_uint(f) + 0x8000u) >> 16;
  union { unsigned short s; __bf16 b; } o; o.s = (unsigned short)u;
  return o.b;
}

__device__ __forceinline__ v16bf cat8(v8bf lo, v8bf hi) {
  v16bf r;
#pragma unroll
  for (int i = 0; i < 8; ++i) { r[i] = lo[i]; r[i + 8] = hi[i]; }
  return r;
}

__device__ __forceinline__ v8f wmma_bf16(v16bf a, v16bf b, v8f c) {
  return __builtin_amdgcn_wmma_f32_16x16x32_bf16(false, a, false, b,
                                                 (short)0, c, false, false);
}

// ---------------------------------------------------------------------------
// Tensor Data Mover: 2-D tile (bf16 elements) global -> LDS.
//   lds_addr:  LDS byte offset (low 32 bits of generic pointer = LDS offset)
//   gptr:      global address of tile start
//   tensor_d0/tensor_d1: tensor extents (elements) for OOB handling
//   tile_d0/tile_d1:     tile extents (elements)
//   stride0:   elements between consecutive dim-1 lines
// Descriptor per CDNA5 ISA §8 (D# group0 128b + group1 256b); groups 2/3 NULL.
// This toolchain's builtin is the 6-arg form:
//   (uint32x4 g0, int32x8 g1, int32x4, int32x4, int32x8, i32 cpol)
// ---------------------------------------------------------------------------
__device__ __forceinline__ void tdm_load_2d(unsigned lds_addr, const void* gptr,
                                            unsigned tensor_d0, unsigned tensor_d1,
                                            unsigned tile_d0, unsigned tile_d1,
                                            unsigned stride0) {
  unsigned long long ga = (unsigned long long)(uintptr_t)gptr;
  v4u g0;
  g0[0] = 1u;                                   // count=1, no gather
  g0[1] = lds_addr;                             // lds_addr [63:32]
  g0[2] = (unsigned)ga;                         // global_addr [95:64]
  g0[3] = (unsigned)((ga >> 32) & 0x1FFFFFFu)   // global_addr [120:96]
        | (2u << 30);                           // type=2 ("image")
  v8i g1;
  g1[0] = (int)(1u << 16);                      // data_size=1 -> 2 bytes
  g1[1] = (int)(tensor_d0 << 16);               // tensor_dim0 [79:48] low
  g1[2] = (int)((tensor_d0 >> 16) | (tensor_d1 << 16));
  g1[3] = (int)((tensor_d1 >> 16) | (tile_d0 << 16));   // tile_dim0 [127:112]
  g1[4] = (int)(tile_d1 & 0xFFFFu);             // tile_dim1, tile_dim2=0
  g1[5] = (int)stride0;                         // tensor_dim0_stride low 32
  g1[6] = 0;
  g1[7] = 0;
  v4i z4 = {0, 0, 0, 0};
  v8i z8 = {0, 0, 0, 0, 0, 0, 0, 0};
  __builtin_amdgcn_tensor_load_to_lds(g0, g1, z4, z4, z8, 0);
}

// ---------------------------------------------------------------------------
// Projection GEMM: one wave per 16(out-ch) x 16(pixel) tile.
// ---------------------------------------------------------------------------
__global__ void proj16(const float* __restrict__ W, const float* __restrict__ bias,
                       const float* __restrict__ x, __bf16* __restrict__ out,
                       int OC, int transpose_out) {
  const int lane = threadIdx.x & 31;
  const int half = lane >> 4;
  const int sub  = lane & 15;
  const int itile = blockIdx.x * 16;
  const int otile = blockIdx.y * 16;
  const int batch = blockIdx.z;
  const int ak0 = half ? 16 : 0;     // A-matrix K chunk offsets (contiguous k = 16*half + s, as B)
  const int ak1 = half ? 24 : 8;
  const int bk  = half ? 16 : 0;     // B-matrix K offset
  const int row_off = half ? 8 : 0;  // C/D row mapping

  v8f acc = {};
  const float* wrow = W + (size_t)(otile + sub) * CIN;
  const float* xcol = x + (size_t)batch * CIN * NPIX + (itile + sub);

  for (int kc = 0; kc < CIN; kc += 32) {
    v8u wu, xu;
#pragma unroll
    for (int t = 0; t < 8; t += 2) {
      wu[t >> 1]       = pkbf(wrow[kc + ak0 + t], wrow[kc + ak0 + t + 1]);
      wu[(t >> 1) + 4] = pkbf(wrow[kc + ak1 + t], wrow[kc + ak1 + t + 1]);
    }
    const float* xc = xcol + (size_t)(kc + bk) * NPIX;
#pragma unroll
    for (int t = 0; t < 16; t += 2)
      xu[t >> 1] = pkbf(xc[(size_t)t * NPIX], xc[(size_t)(t + 1) * NPIX]);
    acc = wmma_bf16(__builtin_bit_cast(v16bf, wu),
                    __builtin_bit_cast(v16bf, xu), acc);
  }

#pragma unroll
  for (int r = 0; r < 8; ++r) acc[r] += bias[otile + row_off + r];

  if (transpose_out) {
    __bf16* p = out + ((size_t)batch * NPIX + itile + sub) * OC + otile + row_off;
    v4u st;
#pragma unroll
    for (int r = 0; r < 8; r += 2) st[r >> 1] = pkbf(acc[r], acc[r + 1]);
    *(v4u*)p = st;      // 8 consecutive bf16 along o -> one 16B store
  } else {
#pragma unroll
    for (int r = 0; r < 8; ++r)
      out[((size_t)batch * OC + otile + row_off + r) * NPIX + itile + sub] =
          f2bf(acc[r]);
  }
}

// ---------------------------------------------------------------------------
// Fused flash attention + residual.
// Block = 256 threads (8 waves) per (batch, 32-query tile).
// Wave w: query subtile (w&1)*16, channels [(w>>1)*64, +64).
// K/V tiles staged to LDS by TDM (wave 0), double buffered.
// ---------------------------------------------------------------------------
__global__ void __launch_bounds__(256, 2)
attn_fused(const __bf16* __restrict__ qT, const __bf16* __restrict__ kT,
           const __bf16* __restrict__ vM, const float* __restrict__ x,
           const float* __restrict__ gamma, float* __restrict__ out) {
  __shared__ __bf16 kbuf[2][32][32];    //  4 KB: 32 keys x 32 d
  __shared__ __bf16 vbuf[2][256][32];   // 32 KB: 256 ch x 32 j
  __shared__ __bf16 plds[8][16][32];    //  8 KB: per-wave P transpose staging

  const int lane = threadIdx.x & 31;
  const int wave = threadIdx.x >> 5;
  const int half = lane >> 4;
  const int sub  = lane & 15;
  const int ak0 = half ? 16 : 0;
  const int ak1 = half ? 24 : 8;
  const int bk  = half ? 16 : 0;
  const int row_off = half ? 8 : 0;
  const int batch = blockIdx.y;
  const int ibase = blockIdx.x * 32 + (wave & 1) * 16;  // 16-query subtile
  const int ch0   = (wave >> 1) * 64;                   // 64-channel group

  // Q tile (16 queries x 32 d) as A-matrix, loaded once.
  const __bf16* qrow = qT + ((size_t)batch * NPIX + ibase + sub) * DQK;
  v16bf qa = cat8(*(const v8bf*)(qrow + ak0), *(const v8bf*)(qrow + ak1));

  v8f o0 = {}, o1 = {}, o2 = {}, o3 = {};
  float mrow[8], lsum[8];
#pragma unroll
  for (int r = 0; r < 8; ++r) { mrow[r] = -3.0e38f; lsum[r] = 0.f; }

  const __bf16* kbase = kT + (size_t)batch * NPIX * DQK;
  const __bf16* vbase = vM + (size_t)batch * CIN * NPIX;

  // Prologue: stage tile 0 into buffer 0 (wave 0 drives the TDM).
  if (wave == 0) {
    tdm_load_2d((unsigned)(uintptr_t)&kbuf[0][0][0], kbase,
                NPIX * DQK, 1u, 32u * DQK, 1u, NPIX * DQK);   // contiguous 2KB
    tdm_load_2d((unsigned)(uintptr_t)&vbuf[0][0][0], vbase,
                NPIX, CIN, 32u, CIN, NPIX);                   // 256 x 64B tile
  }

  const int nsteps = NPIX / 32;   // 128
  for (int jt = 0; jt < nsteps; ++jt) {
    const int buf = jt & 1;
    __syncthreads();   // buf^1 consumers from previous iteration are done

    if (wave == 0) {
      // Always issue the next tile (index wraps; final issue is a harmless
      // dummy) so exactly 4 TDM ops are outstanding -> constant wait imm.
      const int jn = ((jt + 1) & (nsteps - 1)) * 32;
      tdm_load_2d((unsigned)(uintptr_t)&kbuf[buf ^ 1][0][0],
                  kbase + (size_t)jn * DQK,
                  NPIX * DQK, 1u, 32u * DQK, 1u, NPIX * DQK);
      tdm_load_2d((unsigned)(uintptr_t)&vbuf[buf ^ 1][0][0],
                  vbase + jn, NPIX, CIN, 32u, CIN, NPIX);
      __builtin_amdgcn_s_wait_tensorcnt(2);   // current tile pair complete
    }
    __syncthreads();   // staged K/V for this step visible to all waves

    // K B-matrices from LDS: column = key, 16 contiguous d per lane.
    v16bf kb0 = *(const v16bf*)&kbuf[buf][sub][bk];
    v16bf kb1 = *(const v16bf*)&kbuf[buf][16 + sub][bk];
    v8f zero = {};
    v8f s0 = wmma_bf16(qa, kb0, zero);   // S[16q x 16k]
    v8f s1 = wmma_bf16(qa, kb1, zero);

    // --- online softmax across the 16-lane halves (rows r / r+8) ---
    float tm[8];
#pragma unroll
    for (int r = 0; r < 8; ++r) tm[r] = fmaxf(s0[r], s1[r]);
#pragma unroll
    for (int m = 1; m < 16; m <<= 1)
#pragma unroll
      for (int r = 0; r < 8; ++r) tm[r] = fmaxf(tm[r], __shfl_xor(tm[r], m, 32));

    float alpha[8], ts[8];
#pragma unroll
    for (int r = 0; r < 8; ++r) {
      float mn = fmaxf(mrow[r], tm[r]);
      alpha[r] = __expf(mrow[r] - mn);
      mrow[r]  = mn;
      s0[r] = __expf(s0[r] - mn);
      s1[r] = __expf(s1[r] - mn);
      ts[r] = s0[r] + s1[r];
    }
#pragma unroll
    for (int m = 1; m < 16; m <<= 1)
#pragma unroll
      for (int r = 0; r < 8; ++r) ts[r] += __shfl_xor(ts[r], m, 32);
#pragma unroll
    for (int r = 0; r < 8; ++r) {
      lsum[r] = lsum[r] * alpha[r] + ts[r];
      o0[r] *= alpha[r]; o1[r] *= alpha[r];
      o2[r] *= alpha[r]; o3[r] *= alpha[r];
    }

    // --- P: C-layout (row=VGPR, col=lane) -> A-layout via private LDS tile ---
#pragma unroll
    for (int r = 0; r < 8; ++r) {
      plds[wave][row_off + r][sub]      = f2bf(s0[r]);
      plds[wave][row_off + r][16 + sub] = f2bf(s1[r]);
    }
    const __bf16* prow = &plds[wave][sub][0];  // same-wave DS ops stay in order
    v16bf pa = cat8(*(const v8bf*)(prow + ak0), *(const v8bf*)(prow + ak1));

    // V B-matrices from LDS: lane's channel row, 16 contiguous j.
    v16bf vb0 = *(const v16bf*)&vbuf[buf][ch0 + sub][bk];
    v16bf vb1 = *(const v16bf*)&vbuf[buf][ch0 + 16 + sub][bk];
    v16bf vb2 = *(const v16bf*)&vbuf[buf][ch0 + 32 + sub][bk];
    v16bf vb3 = *(const v16bf*)&vbuf[buf][ch0 + 48 + sub][bk];
    o0 = wmma_bf16(pa, vb0, o0);
    o1 = wmma_bf16(pa, vb1, o1);
    o2 = wmma_bf16(pa, vb2, o2);
    o3 = wmma_bf16(pa, vb3, o3);
  }

  // --- epilogue: normalize, gamma-scale, residual; contiguous f32 traffic ---
  const float g = gamma[0];
#pragma unroll
  for (int grp = 0; grp < 4; ++grp) {
    const v8f& oacc = (grp == 0) ? o0 : (grp == 1) ? o1 : (grp == 2) ? o2 : o3;
    size_t base = ((size_t)batch * CIN + ch0 + grp * 16 + sub) * NPIX
                + ibase + row_off;
#pragma unroll
    for (int r = 0; r < 8; ++r)
      out[base + r] = g * oacc[r] / lsum[r] + x[base + r];
  }
}

// ---------------------------------------------------------------------------
extern "C" void kernel_launch(void* const* d_in, const int* in_sizes, int n_in,
                              void* d_out, int out_size, void* d_ws, size_t ws_size,
                              hipStream_t stream) {
  const float* x     = (const float*)d_in[0];
  const float* wq    = (const float*)d_in[1];
  const float* bq    = (const float*)d_in[2];
  const float* wk    = (const float*)d_in[3];
  const float* bk    = (const float*)d_in[4];
  const float* wv    = (const float*)d_in[5];
  const float* bv    = (const float*)d_in[6];
  const float* gamma = (const float*)d_in[7];
  float* out = (float*)d_out;

  // Workspace (bf16): qT [8][4096][32], kT [8][4096][32], v [8][256][4096]
  __bf16* qT = (__bf16*)d_ws;
  __bf16* kT = qT + (size_t)BATCH * NPIX * DQK;
  __bf16* vM = kT + (size_t)BATCH * NPIX * DQK;

  proj16<<<dim3(NPIX / 16, DQK / 16, BATCH), 32, 0, stream>>>(wq, bq, x, qT, DQK, 1);
  proj16<<<dim3(NPIX / 16, DQK / 16, BATCH), 32, 0, stream>>>(wk, bk, x, kT, DQK, 1);
  proj16<<<dim3(NPIX / 16, CIN / 16, BATCH), 32, 0, stream>>>(wv, bv, x, vM, CIN, 0);

  attn_fused<<<dim3(NPIX / 32, BATCH), 256, 0, stream>>>(qT, kT, vM, x, gamma, out);
}
